// DSA_Block_19370302505073
// MI455X (gfx1250) — matrix operand order repaired and checked by reading, not compiled, not run on hardware
//
#include <hip/hip_runtime.h>

#define DEV __device__ __forceinline__

typedef __bf16 bf16;
typedef __attribute__((ext_vector_type(16))) __bf16 v16bf;
typedef __attribute__((ext_vector_type(8)))  float  v8f;

// ---------------- constants ----------------
constexpr int NB = 8;     // batch
constexpr int NC = 256;   // channels
constexpr int NS = 4096;  // spatial 64*64
constexpr int NT = 512;   // temb
constexpr int NK = 128;   // top-k selected channels
constexpr float EPSV = 1e-6f;

// tiled layouts (16x16 tiles of 256 bf16):
//  A-layout: off = ((m>>4)*Ktiles + (k>>4))*256 + (m&15)*16 + (k&15)
//  B-layout: off = ((k>>4)*Ntiles + (n>>4))*256 + (n&15)*16 + (k&15)

DEV float sigmoid_f(float x) { return 1.0f / (1.0f + __expf(-x)); }
DEV float silu_f(float x)    { return x * sigmoid_f(x); }

union FragU { uint4 u[2]; v16bf v; };

// ---------------- kernel 0: temb matvecs ----------------
__global__ void k_temb(const float* __restrict__ temb,
                       const float* tsW, const float* tsB,
                       const float* tpW, const float* tpB,
                       const float* srW, const float* srB,
                       const float* msW, const float* msB,
                       float* spc_ss, float* tp, float* spr_ss, float* msgn_ss) {
  __shared__ float st[NT];
  int b = blockIdx.x, tid = threadIdx.x;
  for (int e = 0; e < 2; ++e) {
    int i = tid + e * 256;
    st[i] = silu_f(temb[b * NT + i]);
  }
  __syncthreads();
  for (int o = tid; o < 1792; o += 256) {
    const float* Wr; float bias; float* dst;
    if (o < 512)       { Wr = tsW + o * NT;            bias = tsB[o];        dst = &spc_ss[b * 512 + o]; }
    else if (o < 768)  { int j = o - 512;  Wr = tpW + j * NT; bias = tpB[j]; dst = &tp[b * 256 + j]; }
    else if (o < 1280) { int j = o - 768;  Wr = srW + j * NT; bias = srB[j]; dst = &spr_ss[b * 512 + j]; }
    else               { int j = o - 1280; Wr = msW + j * NT; bias = msB[j]; dst = &msgn_ss[b * 512 + j]; }
    float acc = bias;
    for (int i = 0; i < NT; ++i) acc += Wr[i] * st[i];
    *dst = acc;
  }
}

// ---------------- groupnorm statistics ----------------
__global__ void k_gnstats(const float* __restrict__ src, float* stats) {
  __shared__ float rs[256], rq[256];
  int bg = blockIdx.x, tid = threadIdx.x;
  long base = (long)bg * 8 * NS;
  float s = 0.f, q = 0.f;
  for (int e = 0; e < 128; ++e) {
    float v = src[base + tid + e * 256];
    s += v; q += v * v;
  }
  rs[tid] = s; rq[tid] = q; __syncthreads();
  for (int w = 128; w > 0; w >>= 1) {
    if (tid < w) { rs[tid] += rs[tid + w]; rq[tid] += rq[tid + w]; }
    __syncthreads();
  }
  if (tid == 0) {
    float mu = rs[0] * (1.f / 32768.f);
    stats[bg * 2]     = mu;
    stats[bg * 2 + 1] = rq[0] * (1.f / 32768.f) - mu * mu;
  }
}

// ---------------- fused gn + time_affine -> tiled-B bf16 (optional GAP) ----------------
__global__ void k_affine(const float* __restrict__ src, const float* stats,
                         const float* gamma, const float* beta, const float* ss,
                         bf16* out, float* gap) {
  __shared__ float red[256];
  int bc = blockIdx.x, tid = threadIdx.x;
  int b = bc >> 8, c = bc & 255, g = c >> 3;
  float mu = stats[(b * 32 + g) * 2], var = stats[(b * 32 + g) * 2 + 1];
  float r  = rsqrtf(var + EPSV);
  float sc = ss[b * 512 + c], sh = ss[b * 512 + 256 + c];
  float A  = gamma[c] * r * (1.f + sc);
  float Bo = (beta[c] - gamma[c] * r * mu) * (1.f + sc) + sh;
  long ibase = (long)bc * NS;
  bf16* ob   = out + (long)b * (16 * 256 * 256) + (long)(c >> 4) * 256 * 256 + (c & 15);
  float s = 0.f;
  for (int e = 0; e < 16; ++e) {
    int p = tid + e * 256;
    float v = src[ibase + p] * A + Bo;
    ob[(long)(p >> 4) * 256 + (p & 15) * 16] = (bf16)v;
    s += v;
  }
  if (gap) {
    red[tid] = s; __syncthreads();
    for (int w = 128; w > 0; w >>= 1) {
      if (tid < w) red[tid] += red[tid + w];
      __syncthreads();
    }
    if (tid == 0) gap[b * 256 + c] = red[0] * (1.f / (float)NS);
  }
}

// ---------------- prompt MLP + top-k(128) ----------------
__global__ void k_prompt(const float* gap, const float* tp,
                         const float* m1W, const float* m1B,
                         const float* m2W, const float* m2B, int* idx) {
  __shared__ float z[512], zz[512], pr[256];
  int b = blockIdx.x, tid = threadIdx.x;
  z[tid]       = gap[b * 256 + tid];
  z[256 + tid] = tp[b * 256 + tid];
  __syncthreads();
  for (int e = 0; e < 2; ++e) {
    int j = tid + e * 256;
    float acc = m1B[j];
    const float* W = m1W + j * 512;
    for (int i = 0; i < 512; ++i) acc += W[i] * z[i];
    zz[j] = silu_f(acc);
  }
  __syncthreads();
  {
    float acc = m2B[tid];
    const float* W = m2W + tid * 512;
    for (int i = 0; i < 512; ++i) acc += W[i] * zz[i];
    pr[tid] = acc;
  }
  __syncthreads();
  float v = pr[tid];
  int cnt = 0;
  for (int j = 0; j < 256; ++j) {
    float u = pr[j];
    cnt += (u > v) || (u == v && j < tid);
  }
  if (cnt < NK) idx[b * NK + cnt] = tid;
}

// ---------------- weight packing (fp32 row-major -> tiled-A bf16) ----------------
__global__ void k_packA(const float* __restrict__ W, bf16* out, int M, int K) {
  long i = (long)blockIdx.x * 256 + threadIdx.x;
  int m = (int)(i / K), k = (int)(i % K);
  int Kt = K >> 4;
  out[((long)(m >> 4) * Kt + (k >> 4)) * 256 + (m & 15) * 16 + (k & 15)] = (bf16)W[i];
}
// gathered (per-batch) q/k/v weights: M=128 selected rows of a 256x256 matrix
__global__ void k_packA_gather(const float* __restrict__ W, const int* idx, bf16* out) {
  long i = (long)blockIdx.x * 256 + threadIdx.x;     // NB*128*256
  int b = (int)(i >> 15);
  int m = (int)((i >> 8) & 127), k = (int)(i & 255);
  int row = idx[b * NK + m];
  out[(long)b * 32768 + ((long)(m >> 4) * 16 + (k >> 4)) * 256 + (m & 15) * 16 + (k & 15)]
      = (bf16)W[row * 256 + k];
}
// conv3 weights: per-tap tiled A (9 x [256x256]); also zero the h3 guard tail
__global__ void k_packA3(const float* __restrict__ W, bf16* out, bf16* htail) {
  long i = (long)blockIdx.x * 256 + threadIdx.x;     // 9*256*256
  if (blockIdx.x == 0) htail[threadIdx.x] = (bf16)0.0f;  // 256 zero elements
  int tap = (int)(i >> 16);
  int m = (int)((i >> 8) & 255), k = (int)(i & 255);
  out[(long)tap * 65536 + ((long)(m >> 4) * 16 + (k >> 4)) * 256 + (m & 15) * 16 + (k & 15)]
      = (bf16)W[(m * 256 + k) * 9 + tap];
}

// ---------------- LDS-free tiled WMMA GEMM (64x256 block tile, 8 waves) ----------------
struct G2 {
  const bf16* At; long a_bs;     // tiled A
  const bf16* Bt; long b_bs;     // tiled B
  int Ktiles, Nt_alloc, Nt_valid;
  const float* bias; const int* gidx; int gN; int bias_gather;
  float* Of32; bf16* Obf; long o_bs;
  int out_N;                     // logical N (row stride for f32 out, store guard)
  int out_mode;                  // -1: f32 row-major; 0: A-tiled; 1: B-tiled transposed; 2: B-tiled
  float scale; int act;          // 0 none, 1 silu, 2 sigmoid
};

__global__ __launch_bounds__(256) void k_gemm2(G2 p) {
  int b = blockIdx.z;
  int tid = threadIdx.x, lane = tid & 31, wave = tid >> 5;
  int mi = wave >> 2, ni = wave & 3;
  int lm = lane & 15, lh = lane >> 4;
  int mt0 = blockIdx.y * 4 + mi * 2;      // this wave owns m-tiles mt0, mt0+1
  int nt_base = blockIdx.x * 16 + ni * 4;

  const uint4* AB = (const uint4*)p.At;   // base pointers (single provenance -> global)
  const uint4* BB = (const uint4*)p.Bt;

  // offsets in uint4 (8 bf16) units
  long oa0 = (long)(b * p.a_bs + (long)mt0 * p.Ktiles * 256) / 8 + lm * 2 + lh * 32;
  long oa1 = oa0 + (long)p.Ktiles * 32;
  long obb = (long)(b * p.b_bs) / 8 + (long)lh * p.Nt_alloc * 32 + lm * 2;
  int nt0c = nt_base + 0, nt1c = nt_base + 1, nt2c = nt_base + 2, nt3c = nt_base + 3;
  if (nt0c >= p.Nt_valid) nt0c = p.Nt_valid - 1;  // clamped: results discarded in epilogue
  if (nt1c >= p.Nt_valid) nt1c = p.Nt_valid - 1;
  if (nt2c >= p.Nt_valid) nt2c = p.Nt_valid - 1;
  if (nt3c >= p.Nt_valid) nt3c = p.Nt_valid - 1;
  long ob0 = obb + (long)nt0c * 32;
  long ob1 = obb + (long)nt1c * 32;
  long ob2 = obb + (long)nt2c * 32;
  long ob3 = obb + (long)nt3c * 32;
  long bstep = (long)p.Nt_alloc * 64;  // uint4 per (kt += 2)

  v8f acc[2][4];
#pragma unroll
  for (int t = 0; t < 2; ++t)
#pragma unroll
    for (int j = 0; j < 4; ++j)
#pragma unroll
      for (int r = 0; r < 8; ++r) acc[t][j][r] = 0.0f;

  for (int kt = 0; kt < p.Ktiles; kt += 2) {
    FragU ua0, ua1, ub;
    ua0.u[0] = AB[oa0]; ua0.u[1] = AB[oa0 + 1];
    ua1.u[0] = AB[oa1]; ua1.u[1] = AB[oa1 + 1];
    __builtin_prefetch((const void*)(AB + oa0 + 64), 0, 3);
    oa0 += 64; oa1 += 64;

    ub.u[0] = BB[ob0]; ub.u[1] = BB[ob0 + 1]; ob0 += bstep;
    acc[0][0] = __builtin_amdgcn_wmma_f32_16x16x32_bf16(false, ua0.v, false, ub.v, (short)0, acc[0][0], false, false);
    acc[1][0] = __builtin_amdgcn_wmma_f32_16x16x32_bf16(false, ua1.v, false, ub.v, (short)0, acc[1][0], false, false);

    ub.u[0] = BB[ob1]; ub.u[1] = BB[ob1 + 1]; ob1 += bstep;
    acc[0][1] = __builtin_amdgcn_wmma_f32_16x16x32_bf16(false, ua0.v, false, ub.v, (short)0, acc[0][1], false, false);
    acc[1][1] = __builtin_amdgcn_wmma_f32_16x16x32_bf16(false, ua1.v, false, ub.v, (short)0, acc[1][1], false, false);

    ub.u[0] = BB[ob2]; ub.u[1] = BB[ob2 + 1]; ob2 += bstep;
    acc[0][2] = __builtin_amdgcn_wmma_f32_16x16x32_bf16(false, ua0.v, false, ub.v, (short)0, acc[0][2], false, false);
    acc[1][2] = __builtin_amdgcn_wmma_f32_16x16x32_bf16(false, ua1.v, false, ub.v, (short)0, acc[1][2], false, false);

    ub.u[0] = BB[ob3]; ub.u[1] = BB[ob3 + 1]; ob3 += bstep;
    acc[0][3] = __builtin_amdgcn_wmma_f32_16x16x32_bf16(false, ua0.v, false, ub.v, (short)0, acc[0][3], false, false);
    acc[1][3] = __builtin_amdgcn_wmma_f32_16x16x32_bf16(false, ua1.v, false, ub.v, (short)0, acc[1][3], false, false);
  }

#pragma unroll
  for (int t = 0; t < 2; ++t) {
    int m0 = (mt0 + t) * 16 + 8 * lh;
#pragma unroll
    for (int j = 0; j < 4; ++j) {
      int n = (nt_base + j) * 16 + lm;
      if (n >= p.out_N) continue;
#pragma unroll
      for (int r = 0; r < 8; ++r) {
        int m = m0 + r;
        float v = acc[t][j][r] * p.scale;
        if (p.bias) {
          int bi = p.bias_gather ? p.gidx[b * p.gN + m] : m;
          v += p.bias[bi];
        }
        if (p.act == 1) v = silu_f(v);
        else if (p.act == 2) v = sigmoid_f(v);
        if (p.out_mode < 0) {
          p.Of32[(long)b * p.o_bs + (long)m * p.out_N + n] = v;
        } else {
          long off;
          if (p.out_mode == 0)      // A-tiled (M x N)
            off = ((long)(m >> 4) * (p.out_N >> 4) + (n >> 4)) * 256 + (m & 15) * 16 + (n & 15);
          else if (p.out_mode == 1) // B-tiled transposed: K=n-dim, N=m-dim (8 row-tiles)
            off = ((long)(n >> 4) * 8 + (m >> 4)) * 256 + (m & 15) * 16 + (n & 15);
          else                      // B-tiled: K=m-dim, N=n-dim
            off = ((long)(m >> 4) * (p.out_N >> 4) + (n >> 4)) * 256 + (n & 15) * 16 + (m & 15);
          p.Obf[(long)b * p.o_bs + off] = (bf16)v;
        }
      }
    }
  }
}

// ---------------- 3x3 conv: implicit WMMA GEMM, direct tiled loads ----------------
// H has a zeroed 256-element tail at element offset NB*1048576 (border guard)
__global__ __launch_bounds__(256) void k_conv3(const bf16* __restrict__ Apack,
                                               const float* bias,
                                               const bf16* __restrict__ H,
                                               float* Out) {
  int b = blockIdx.z;
  int tid = threadIdx.x, lane = tid & 31, wave = tid >> 5;
  int mi = wave >> 2, ni = wave & 3;
  int lm = lane & 15, lh = lane >> 4;
  int mt0 = blockIdx.y * 4 + mi * 2;
  const uint4* AB = (const uint4*)Apack;
  const uint4* HB = (const uint4*)H;
  const long zoff = (long)NB * 131072;   // zero tail, uint4 units

  v8f acc[2][4];
#pragma unroll
  for (int t = 0; t < 2; ++t)
#pragma unroll
    for (int j = 0; j < 4; ++j)
#pragma unroll
      for (int r = 0; r < 8; ++r) acc[t][j][r] = 0.0f;

  for (int tap = 0; tap < 9; ++tap) {
    int dy = tap / 3 - 1, dx = tap % 3 - 1;
    long oa0 = (long)tap * 8192 + (long)mt0 * 16 * 32 + lm * 2 + lh * 32;
    long oa1 = oa0 + 16 * 32;
    long ob0, ob1, ob2, ob3, s0, s1, s2, s3;
    long hb = (long)b * 131072 + (long)lh * 256 * 32;
#pragma unroll
    for (int j = 0; j < 4; ++j) {
      int n = blockIdx.x * 256 + ni * 64 + j * 16 + lm;
      int y = (n >> 6) + dy, x = (n & 63) + dx;
      bool ok = ((unsigned)y < 64u) && ((unsigned)x < 64u);
      int pp = ok ? ((y << 6) + x) : 0;
      long o = ok ? (hb + (long)(pp >> 4) * 32 + (pp & 15) * 2) : zoff;
      long s = ok ? 16384 : 0;   // +2 K-tiles of 256*256 bf16 (in uint4)
      if (j == 0) { ob0 = o; s0 = s; }
      else if (j == 1) { ob1 = o; s1 = s; }
      else if (j == 2) { ob2 = o; s2 = s; }
      else { ob3 = o; s3 = s; }
    }
    for (int kt = 0; kt < 16; kt += 2) {
      FragU ua0, ua1, ub;
      ua0.u[0] = AB[oa0]; ua0.u[1] = AB[oa0 + 1];
      ua1.u[0] = AB[oa1]; ua1.u[1] = AB[oa1 + 1];
      oa0 += 64; oa1 += 64;

      ub.u[0] = HB[ob0]; ub.u[1] = HB[ob0 + 1]; ob0 += s0;
      acc[0][0] = __builtin_amdgcn_wmma_f32_16x16x32_bf16(false, ua0.v, false, ub.v, (short)0, acc[0][0], false, false);
      acc[1][0] = __builtin_amdgcn_wmma_f32_16x16x32_bf16(false, ua1.v, false, ub.v, (short)0, acc[1][0], false, false);

      ub.u[0] = HB[ob1]; ub.u[1] = HB[ob1 + 1]; ob1 += s1;
      acc[0][1] = __builtin_amdgcn_wmma_f32_16x16x32_bf16(false, ua0.v, false, ub.v, (short)0, acc[0][1], false, false);
      acc[1][1] = __builtin_amdgcn_wmma_f32_16x16x32_bf16(false, ua1.v, false, ub.v, (short)0, acc[1][1], false, false);

      ub.u[0] = HB[ob2]; ub.u[1] = HB[ob2 + 1]; ob2 += s2;
      acc[0][2] = __builtin_amdgcn_wmma_f32_16x16x32_bf16(false, ua0.v, false, ub.v, (short)0, acc[0][2], false, false);
      acc[1][2] = __builtin_amdgcn_wmma_f32_16x16x32_bf16(false, ua1.v, false, ub.v, (short)0, acc[1][2], false, false);

      ub.u[0] = HB[ob3]; ub.u[1] = HB[ob3 + 1]; ob3 += s3;
      acc[0][3] = __builtin_amdgcn_wmma_f32_16x16x32_bf16(false, ua0.v, false, ub.v, (short)0, acc[0][3], false, false);
      acc[1][3] = __builtin_amdgcn_wmma_f32_16x16x32_bf16(false, ua1.v, false, ub.v, (short)0, acc[1][3], false, false);
    }
  }
#pragma unroll
  for (int t = 0; t < 2; ++t) {
    int m0 = (mt0 + t) * 16 + 8 * lh;
#pragma unroll
    for (int j = 0; j < 4; ++j) {
      int n = blockIdx.x * 256 + ni * 64 + j * 16 + lm;
#pragma unroll
      for (int r = 0; r < 8; ++r) {
        int m = m0 + r;
        Out[((long)b * NC + m) * NS + n] = acc[t][j][r] + bias[m];
      }
    }
  }
}

// ---------------- depthwise 3x3 (fused gn+affine, silu) -> tiled-B bf16 ----------------
__global__ void k_dw(const float* __restrict__ x, const float* stats,
                     const float* gamma, const float* beta, const float* ss,
                     const float* dww, const float* dwb, bf16* out) {
  __shared__ float tile[NS];
  int bc = blockIdx.x, tid = threadIdx.x;
  int b = bc >> 8, c = bc & 255, g = c >> 3;
  float mu = stats[(b * 32 + g) * 2], var = stats[(b * 32 + g) * 2 + 1];
  float r  = rsqrtf(var + EPSV);
  float sc = ss[b * 512 + c], sh = ss[b * 512 + 256 + c];
  float A  = gamma[c] * r * (1.f + sc);
  float Bo = (beta[c] - gamma[c] * r * mu) * (1.f + sc) + sh;
  long ibase = (long)bc * NS;
  for (int e = 0; e < 16; ++e) {
    int p = tid + e * 256;
    tile[p] = x[ibase + p] * A + Bo;
  }
  __syncthreads();
  float w[9];
#pragma unroll
  for (int t = 0; t < 9; ++t) w[t] = dww[c * 9 + t];
  float bb = dwb[c];
  bf16* ob = out + (long)b * 1048576 + (long)(c >> 4) * 256 * 256 + (c & 15);
  for (int e = 0; e < 16; ++e) {
    int p = tid + e * 256;
    int y = p >> 6, xx = p & 63;
    float s = bb;
#pragma unroll
    for (int t = 0; t < 9; ++t) {
      int yy = y + t / 3 - 1, xc = xx + t % 3 - 1;
      if ((unsigned)yy < 64u && (unsigned)xc < 64u)
        s += w[t] * tile[(yy << 6) + xc];
    }
    ob[(long)(p >> 4) * 256 + (p & 15) * 16] = (bf16)silu_f(s);
  }
}

// ---------------- softmax over 128 ----------------
__global__ void k_softmax(const float* __restrict__ S, bf16* P) {
  __shared__ float red[128];
  int row = blockIdx.x, tid = threadIdx.x;
  long off = (long)row * 128 + tid;
  float v = S[off];
  red[tid] = v; __syncthreads();
  for (int w = 64; w > 0; w >>= 1) {
    if (tid < w) red[tid] = fmaxf(red[tid], red[tid + w]);
    __syncthreads();
  }
  float mx = red[0]; __syncthreads();
  float e = __expf(v - mx);
  red[tid] = e; __syncthreads();
  for (int w = 64; w > 0; w >>= 1) {
    if (tid < w) red[tid] += red[tid + w];
    __syncthreads();
  }
  P[off] = (bf16)(e / red[0]);
}

// ---------------- W2 = Wproj[:, idx] @ P  -> tiled-A bf16 (M=256,K=128) ----------------
__global__ void k_w2(const float* __restrict__ Wproj, const int* idx,
                     const bf16* __restrict__ P, bf16* W2) {
  __shared__ float wl[128];
  int blk = blockIdx.x, tid = threadIdx.x;
  int b = blk >> 8, m = blk & 255;
  wl[tid] = Wproj[m * 256 + idx[b * NK + tid]];
  __syncthreads();
  float acc = 0.f;
  for (int l = 0; l < 128; ++l)
    acc += wl[l] * (float)P[((long)b * NK + l) * 128 + tid];
  int j = tid;
  W2[(long)b * 32768 + ((long)(m >> 4) * 8 + (j >> 4)) * 256 + (m & 15) * 16 + (j & 15)]
      = (bf16)acc;
}

// ---------------- elementwise combines ----------------
__global__ void k_combine(const float* x, const float* fg, const float* fl,
                          const float* w1r, const float* w2r, float* x1) {
  long i = (long)blockIdx.x * 256 + threadIdx.x;
  int c = (int)((i >> 12) & 255);
  x1[i] = x[i] + w1r[c] * fg[i] + w2r[c] * fl[i];
}
__global__ void k_final(const float* x1, const float* a, const float* gate, float* out) {
  long i = (long)blockIdx.x * 256 + threadIdx.x;
  out[i] = x1[i] + a[i] * gate[i];
}

// ================================================================
extern "C" void kernel_launch(void* const* d_in, const int* in_sizes, int n_in,
                              void* d_out, int out_size, void* d_ws, size_t ws_size,
                              hipStream_t stream) {
  (void)in_sizes; (void)n_in; (void)out_size; (void)ws_size;
  const float* x         = (const float*)d_in[0];
  const float* temb      = (const float*)d_in[1];
  const float* ng_g      = (const float*)d_in[2];
  const float* ng_b      = (const float*)d_in[3];
  const float* nl_g      = (const float*)d_in[4];
  const float* nl_b      = (const float*)d_in[5];
  const float* nf_g      = (const float*)d_in[6];
  const float* nf_b      = (const float*)d_in[7];
  const float* spc_ts_W  = (const float*)d_in[8];
  const float* spc_ts_b  = (const float*)d_in[9];
  const float* spc_tp_W  = (const float*)d_in[10];
  const float* spc_tp_b  = (const float*)d_in[11];
  const float* spc_m1_W  = (const float*)d_in[12];
  const float* spc_m1_b  = (const float*)d_in[13];
  const float* spc_m2_W  = (const float*)d_in[14];
  const float* spc_m2_b  = (const float*)d_in[15];
  const float* spc_q_w   = (const float*)d_in[16];
  const float* spc_q_b   = (const float*)d_in[17];
  const float* spc_k_w   = (const float*)d_in[18];
  const float* spc_k_b   = (const float*)d_in[19];
  const float* spc_v_w   = (const float*)d_in[20];
  const float* spc_v_b   = (const float*)d_in[21];
  const float* spc_pj_w  = (const float*)d_in[22];
  const float* spc_pj_b  = (const float*)d_in[23];
  const float* spr_ts_W  = (const float*)d_in[24];
  const float* spr_ts_b  = (const float*)d_in[25];
  const float* spr_dw_w  = (const float*)d_in[26];
  const float* spr_dw_b  = (const float*)d_in[27];
  const float* spr_pw_w  = (const float*)d_in[28];
  const float* spr_pw_b  = (const float*)d_in[29];
  const float* aafm_w1r  = (const float*)d_in[30];
  const float* aafm_w2r  = (const float*)d_in[32];
  const float* msgn_ts_W = (const float*)d_in[34];
  const float* msgn_ts_b = (const float*)d_in[35];
  const float* msgn_a_w  = (const float*)d_in[36];
  const float* msgn_a_b  = (const float*)d_in[37];
  const float* msgn_b_w  = (const float*)d_in[38];
  const float* msgn_b_b  = (const float*)d_in[39];
  float* out = (float*)d_out;

  char* wp = (char*)d_ws;
  auto alloc = [&](size_t n) -> char* {
    char* p = wp; wp += (n + 255) & ~(size_t)255; return p;
  };
  const long BCS = (long)NB * NC * NS;   // 8.39M elems
  float* spc_ss  = (float*)alloc(NB * 512 * 4);
  float* tp      = (float*)alloc(NB * 256 * 4);
  float* spr_ss  = (float*)alloc(NB * 512 * 4);
  float* msgn_ss = (float*)alloc(NB * 512 * 4);
  float* stats0  = (float*)alloc(NB * 64 * 4);
  float* stats1  = (float*)alloc(NB * 64 * 4);
  float* gap     = (float*)alloc(NB * 256 * 4);
  int*   idx     = (int*)  alloc(NB * NK * 4);
  float* Slog    = (float*)alloc((long)NB * NK * 128 * 4);
  bf16*  P       = (bf16*) alloc((long)NB * NK * 128 * 2);
  bf16*  W2t     = (bf16*) alloc((long)NB * 32768 * 2);
  bf16*  h1t     = (bf16*) alloc(BCS * 2);
  bf16*  h3t     = (bf16*) alloc((BCS + 256) * 2);   // +256 zeroed guard tail
  bf16*  dwot    = (bf16*) alloc(BCS * 2);
  bf16*  qgt     = (bf16*) alloc((long)NB * 524288 * 2);
  bf16*  kgt     = (bf16*) alloc((long)NB * 524288 * 2);
  bf16*  vgt     = (bf16*) alloc((long)NB * 524288 * 2);
  bf16*  qA      = (bf16*) alloc((long)NB * 32768 * 2);
  bf16*  kA      = (bf16*) alloc((long)NB * 32768 * 2);
  bf16*  vA      = (bf16*) alloc((long)NB * 32768 * 2);
  bf16*  pwA     = (bf16*) alloc(65536 * 2);
  bf16*  mbA     = (bf16*) alloc(65536 * 2);
  bf16*  c3A     = (bf16*) alloc((long)9 * 65536 * 2);
  float* fg      = (float*)alloc(BCS * 4);
  float* fl      = (float*)alloc(BCS * 4);
  float* x1      = (float*)alloc(BCS * 4);
  float* abuf    = (float*)alloc(BCS * 4);
  float* gatebuf = (float*)alloc(BCS * 4);

  // ---- scalars/weights prep ----
  k_temb<<<NB, 256, 0, stream>>>(temb, spc_ts_W, spc_ts_b, spc_tp_W, spc_tp_b,
                                 spr_ts_W, spr_ts_b, msgn_ts_W, msgn_ts_b,
                                 spc_ss, tp, spr_ss, msgn_ss);
  k_packA<<<256, 256, 0, stream>>>(spr_pw_w, pwA, 256, 256);
  k_packA<<<256, 256, 0, stream>>>(msgn_b_w, mbA, 256, 256);
  k_packA3<<<2304, 256, 0, stream>>>(msgn_a_w, c3A, h3t + BCS);

  // ---- spc branch ----
  k_gnstats<<<NB * 32, 256, 0, stream>>>(x, stats0);
  k_affine<<<NB * NC, 256, 0, stream>>>(x, stats0, ng_g, ng_b, spc_ss, h1t, gap);
  k_prompt<<<NB, 256, 0, stream>>>(gap, tp, spc_m1_W, spc_m1_b, spc_m2_W, spc_m2_b, idx);
  k_packA_gather<<<1024, 256, 0, stream>>>(spc_q_w, idx, qA);
  k_packA_gather<<<1024, 256, 0, stream>>>(spc_k_w, idx, kA);
  k_packA_gather<<<1024, 256, 0, stream>>>(spc_v_w, idx, vA);

  G2 g{};
  g.a_bs = 32768; g.Bt = h1t; g.b_bs = BCS / NB; g.Ktiles = 16;
  g.Nt_alloc = 256; g.Nt_valid = 256;
  g.gidx = idx; g.gN = NK; g.bias_gather = 1;
  g.o_bs = 524288; g.out_N = NS; g.scale = 1.f; g.act = 0; g.Of32 = nullptr;
  g.At = qA; g.bias = spc_q_b; g.Obf = qgt; g.out_mode = 0;   // qg -> A-tiled
  k_gemm2<<<dim3(16, 2, NB), 256, 0, stream>>>(g);
  g.At = kA; g.bias = spc_k_b; g.Obf = kgt; g.out_mode = 1;   // kg -> B-tiled(T)
  k_gemm2<<<dim3(16, 2, NB), 256, 0, stream>>>(g);
  g.At = vA; g.bias = spc_v_b; g.Obf = vgt; g.out_mode = 2;   // vg -> B-tiled
  k_gemm2<<<dim3(16, 2, NB), 256, 0, stream>>>(g);

  // ---- spr branch ----
  k_dw<<<NB * NC, 256, 0, stream>>>(x, stats0, nl_g, nl_b, spr_ss, spr_dw_w, spr_dw_b, dwot);
  G2 gp{};
  gp.At = pwA; gp.a_bs = 0; gp.Bt = dwot; gp.b_bs = BCS / NB; gp.Ktiles = 16;
  gp.Nt_alloc = 256; gp.Nt_valid = 256;
  gp.bias = spr_pw_b; gp.bias_gather = 0;
  gp.Of32 = fl; gp.o_bs = BCS / NB; gp.out_N = NS; gp.out_mode = -1;
  gp.scale = 1.f; gp.act = 1;
  k_gemm2<<<dim3(16, 4, NB), 256, 0, stream>>>(gp);

  // ---- attention ----
  G2 ga{};
  ga.At = qgt; ga.a_bs = 524288; ga.Bt = kgt; ga.b_bs = 524288; ga.Ktiles = 256;
  ga.Nt_alloc = 8; ga.Nt_valid = 8;
  ga.bias = nullptr; ga.bias_gather = 0;
  ga.Of32 = Slog; ga.o_bs = (long)NK * 128; ga.out_N = 128; ga.out_mode = -1;
  ga.scale = rsqrtf((float)NK); ga.act = 0;
  k_gemm2<<<dim3(1, 2, NB), 256, 0, stream>>>(ga);

  k_softmax<<<NB * NK, 128, 0, stream>>>(Slog, P);
  k_w2<<<NB * NC, 128, 0, stream>>>(spc_pj_w, idx, P, W2t);

  G2 go{};
  go.At = W2t; go.a_bs = 32768; go.Bt = vgt; go.b_bs = 524288; go.Ktiles = 8;
  go.Nt_alloc = 256; go.Nt_valid = 256;
  go.bias = spc_pj_b; go.bias_gather = 0;
  go.Of32 = fg; go.o_bs = BCS / NB; go.out_N = NS; go.out_mode = -1;
  go.scale = 1.f; go.act = 0;
  k_gemm2<<<dim3(16, 4, NB), 256, 0, stream>>>(go);

  // ---- AAFM (exact: scalar spectral weights) + residual ----
  k_combine<<<(unsigned)(BCS / 256), 256, 0, stream>>>(x, fg, fl, aafm_w1r, aafm_w2r, x1);

  // ---- msgn branch ----
  k_gnstats<<<NB * 32, 256, 0, stream>>>(x1, stats1);
  k_affine<<<NB * NC, 256, 0, stream>>>(x1, stats1, nf_g, nf_b, msgn_ss, h3t, nullptr);

  G2 gb{};
  gb.At = mbA; gb.a_bs = 0; gb.Bt = h3t; gb.b_bs = BCS / NB; gb.Ktiles = 16;
  gb.Nt_alloc = 256; gb.Nt_valid = 256;
  gb.bias = msgn_b_b; gb.bias_gather = 0;
  gb.Of32 = gatebuf; gb.o_bs = BCS / NB; gb.out_N = NS; gb.out_mode = -1;
  gb.scale = 1.f; gb.act = 2;  // sigmoid gate
  k_gemm2<<<dim3(16, 4, NB), 256, 0, stream>>>(gb);

  k_conv3<<<dim3(16, 4, NB), 256, 0, stream>>>(c3A, msgn_a_b, h3t, abuf);

  k_final<<<(unsigned)(BCS / 256), 256, 0, stream>>>(x1, abuf, gatebuf, out);
}
